// SinkhornDistance_51479478010352
// MI455X (gfx1250) — matrix operand order repaired and checked by reading, not compiled, not run on hardware
//
#include <hip/hip_runtime.h>
#include <cmath>

// ---------------------------------------------------------------------------
// Sinkhorn distance for MI455X (gfx1250, wave32, WMMA + async-to-LDS).
//   B=4, N=M=1024, D=512.
//   Stage 1: norms + f32->f16 convert.
//   Stage 2: C = 1 - cos(x,y) via v_wmma_f32_16x16x32_f16; tiles fed through
//            GLOBAL_LOAD_ASYNC_TO_LDS_B128 (ASYNCcnt double-buffer pipeline).
//            Epilogue writes C and C^T so both Sinkhorn half-steps are
//            coalesced wave-per-row logsumexp sweeps (4096 waves each).
//   Stage 3: 20 Sinkhorn iterations with device-side convergence freeze.
//   Stage 4: cost[b] = sum exp((u+v-C)/eps)*C  (row partials + tree reduce).
// ---------------------------------------------------------------------------

#define B_ 4
#define N_ 1024
#define M_ 1024
#define D_ 512
#define EPS_ 0.1f
#define INV_EPS_ 10.0f
#define MAX_ITER_ 20
#define THRESH_ 0.1f

typedef _Float16 half8  __attribute__((ext_vector_type(8)));
typedef _Float16 half16 __attribute__((ext_vector_type(16)));
typedef float    f32x8  __attribute__((ext_vector_type(8)));

// Async copy of 16 bytes global -> LDS, tracked by ASYNCcnt (gfx1250 TDM-lite
// path). vdst = 32-bit LDS byte address, vaddr = 64-bit global address.
__device__ __forceinline__ void async_b128(unsigned lds_off,
                                           const void* gptr) {
  asm volatile("global_load_async_to_lds_b128 %0, %1, off"
               :: "v"(lds_off), "v"((unsigned long long)(size_t)gptr)
               : "memory");
}

// -------------------------------- init ------------------------------------
__global__ void init_state(float* __restrict__ u, float* __restrict__ v,
                           int* __restrict__ done) {
  int i = blockIdx.x * blockDim.x + threadIdx.x;
  if (i < B_ * N_) u[i] = 0.0f;
  if (i < B_ * M_) v[i] = 0.0f;
  if (i == 0) *done = 0;
}

// ------------------------- norms + f16 convert -----------------------------
__global__ void norm_convert(const float* __restrict__ src,
                             _Float16* __restrict__ dst,
                             float* __restrict__ norms, int rows) {
  int wid = threadIdx.x >> 5, lane = threadIdx.x & 31;
  int row = blockIdx.x * 8 + wid;
  if (row >= rows) return;
  const float* s = src + (size_t)row * D_;
  _Float16* d = dst + (size_t)row * D_;
  float acc = 0.0f;
  for (int k = lane; k < D_; k += 32) {
    float val = s[k];
    acc += val * val;
    d[k] = (_Float16)val;
  }
  for (int off = 16; off; off >>= 1) acc += __shfl_xor(acc, off, 32);
  if (lane == 0) norms[row] = sqrtf(acc);
}

// ------------------------- WMMA GEMM + cost epilogue -----------------------
// Block: 64x64 tile, 8 waves (4x2), each wave two 16x16 f32 accumulators.
// K-tile 32, double-buffered LDS fed by async-to-LDS loads.
#define LDT 40  // padded LDS row stride in halfs (80B -> conflict-free b128)
#define KT_ (D_ / 32)

__global__ __launch_bounds__(256) void gemm_cost(
    const _Float16* __restrict__ Xh, const _Float16* __restrict__ Yh,
    const float* __restrict__ nx, const float* __restrict__ ny,
    float* __restrict__ C, float* __restrict__ CT) {
  __shared__ __align__(16) _Float16 sA[2][64 * LDT];
  __shared__ __align__(16) _Float16 sB[2][64 * LDT];

  const int b   = blockIdx.y;
  const int tm  = blockIdx.x >> 4;   // x-row tile (N/64 = 16)
  const int tn  = blockIdx.x & 15;   // y-row tile (M/64 = 16)
  const int tid = threadIdx.x;
  const int wid = tid >> 5, lane = tid & 31;
  const int wm = wid >> 1, wn = wid & 1;
  const int r = lane & 15, h = lane >> 4;

  const _Float16* Xb = Xh + (size_t)b * N_ * D_;
  const _Float16* Yb = Yh + (size_t)b * M_ * D_;

  f32x8 acc0 = {};
  f32x8 acc1 = {};

  const int lrow = tid >> 2;        // 0..63
  const int lseg = (tid & 3) * 8;   // 0,8,16,24
  const unsigned ldsA0 = (unsigned)(size_t)&sA[0][lrow * LDT + lseg];
  const unsigned ldsA1 = (unsigned)(size_t)&sA[1][lrow * LDT + lseg];
  const unsigned ldsB0 = (unsigned)(size_t)&sB[0][lrow * LDT + lseg];
  const unsigned ldsB1 = (unsigned)(size_t)&sB[1][lrow * LDT + lseg];
  const _Float16* gA = &Xb[(tm * 64 + lrow) * D_ + lseg];
  const _Float16* gB = &Yb[(tn * 64 + lrow) * D_ + lseg];

  // Prologue: async-load tile 0 into buffer 0.
  async_b128(ldsA0, gA);
  async_b128(ldsB0, gB);

  for (int kt = 0; kt < KT_; ++kt) {
    const int cur = kt & 1;
    __syncthreads();  // all waves done reading buf[cur^1] (iteration kt-1)
    if (kt + 1 < KT_) {
      const int koff = (kt + 1) * 32;
      async_b128(cur ? ldsA0 : ldsA1, gA + koff);
      async_b128(cur ? ldsB0 : ldsB1, gB + koff);
      // 2 newer async loads in flight; async loads complete in order, so
      // asynccnt<=2 means tile kt has fully landed in LDS (for this wave).
      asm volatile("s_wait_asynccnt 0x2" ::: "memory");
    } else {
      asm volatile("s_wait_asynccnt 0x0" ::: "memory");
    }
    __syncthreads();  // every wave's tile-kt data visible in LDS

    // A fragment (16x32 f16): lane half h holds contiguous K = 16h..16h+15.
    const _Float16* ap = &sA[cur][(wm * 16 + r) * LDT];
    half8 a_lo = *(const half8*)&ap[16 * h];
    half8 a_hi = *(const half8*)&ap[16 * h + 8];
    half16 afrag = __builtin_shufflevector(a_lo, a_hi, 0, 1, 2, 3, 4, 5, 6, 7,
                                           8, 9, 10, 11, 12, 13, 14, 15);
    // B fragment (32x16 f16): lane half h holds contiguous K = 16h..16h+15.
    const _Float16* bp0 = &sB[cur][(wn * 32 + r) * LDT + 16 * h];
    half16 bfrag0 = __builtin_shufflevector(
        *(const half8*)bp0, *(const half8*)(bp0 + 8), 0, 1, 2, 3, 4, 5, 6, 7,
        8, 9, 10, 11, 12, 13, 14, 15);
    const _Float16* bp1 = &sB[cur][(wn * 32 + 16 + r) * LDT + 16 * h];
    half16 bfrag1 = __builtin_shufflevector(
        *(const half8*)bp1, *(const half8*)(bp1 + 8), 0, 1, 2, 3, 4, 5, 6, 7,
        8, 9, 10, 11, 12, 13, 14, 15);

    acc0 = __builtin_amdgcn_wmma_f32_16x16x32_f16(false, afrag, false, bfrag0,
                                                  (short)0, acc0, false, false);
    acc1 = __builtin_amdgcn_wmma_f32_16x16x32_f16(false, afrag, false, bfrag1,
                                                  (short)0, acc1, false, false);
  }

  // Epilogue: C layout => row = 8*h + vgpr_idx, col = lane%16.
  const int gcol0 = tn * 64 + wn * 32 + r;
  const int gcol1 = gcol0 + 16;
  const float ny0 = ny[b * M_ + gcol0];
  const float ny1 = ny[b * M_ + gcol1];
  float* Cb  = C  + (size_t)b * N_ * M_;
  float* CTb = CT + (size_t)b * M_ * N_;
  for (int i = 0; i < 8; ++i) {
    const int grow = tm * 64 + wm * 16 + h * 8 + i;
    const float nxv = nx[b * N_ + grow];
    const float d0 = 1.0f - acc0[i] / fmaxf(nxv * ny0, 1e-8f);
    const float d1 = 1.0f - acc1[i] / fmaxf(nxv * ny1, 1e-8f);
    Cb[(size_t)grow * M_ + gcol0] = d0;
    Cb[(size_t)grow * M_ + gcol1] = d1;
    CTb[(size_t)gcol0 * N_ + grow] = d0;  // transpose for the v half-step
    CTb[(size_t)gcol1 * N_ + grow] = d1;
  }
}

// ---------------------- Sinkhorn u-update (row lse) ------------------------
// u_new = eps*(log_mu - lse_m((v_m - C_nm)/eps))   [u terms cancel exactly]
__global__ void sink_u(const float* __restrict__ C, const float* __restrict__ v,
                       float* __restrict__ u, float* __restrict__ err_rows,
                       const int* __restrict__ done) {
  if (*done) return;
  const int wid = threadIdx.x >> 5, lane = threadIdx.x & 31;
  const int row = blockIdx.x * 8 + wid;  // b*N + n
  const int b = row >> 10;
  const float* Crow = C + (size_t)row * M_;
  const float* vb = v + b * M_;
  float m = -INFINITY, s = 0.0f;
  for (int j = lane; j < M_; j += 32) {
    float z = (vb[j] - Crow[j]) * INV_EPS_;
    if (z <= m) {
      s += expf(z - m);
    } else {
      s = s * expf(m - z) + 1.0f;
      m = z;
    }
  }
  for (int off = 16; off; off >>= 1) {
    float m2 = __shfl_xor(m, off, 32);
    float s2 = __shfl_xor(s, off, 32);
    float nm = fmaxf(m, m2);
    s = s * expf(m - nm) + s2 * expf(m2 - nm);
    m = nm;
  }
  if (lane == 0) {
    float lse = m + logf(s);
    float un = EPS_ * (logf(1.0f / N_ + 1e-8f) - lse);
    err_rows[row] = fabsf(un - u[row]);
    u[row] = un;
  }
}

// ---------------- Sinkhorn v-update (row lse over C^T) ---------------------
// v_new = eps*(log_nu - lse_n((u_n - C_nm)/eps)); wave-per-row on CT.
__global__ void sink_v(const float* __restrict__ CT,
                       const float* __restrict__ u,
                       const float* __restrict__ nu, float* __restrict__ v,
                       const int* __restrict__ done) {
  if (*done) return;
  const int wid = threadIdx.x >> 5, lane = threadIdx.x & 31;
  const int row = blockIdx.x * 8 + wid;  // b*M + m
  const int b = row >> 10;
  const float* Crow = CT + (size_t)row * N_;
  const float* ub = u + b * N_;
  float m = -INFINITY, s = 0.0f;
  for (int j = lane; j < N_; j += 32) {
    float z = (ub[j] - Crow[j]) * INV_EPS_;
    if (z <= m) {
      s += expf(z - m);
    } else {
      s = s * expf(m - z) + 1.0f;
      m = z;
    }
  }
  for (int off = 16; off; off >>= 1) {
    float m2 = __shfl_xor(m, off, 32);
    float s2 = __shfl_xor(s, off, 32);
    float nm = fmaxf(m, m2);
    s = s * expf(m - nm) + s2 * expf(m2 - nm);
    m = nm;
  }
  if (lane == 0) {
    float lse = m + logf(s);
    v[row] = EPS_ * (logf(nu[row] + 1e-8f) - lse);
  }
}

// -------------------- convergence check (deterministic) --------------------
__global__ void sink_done(const float* __restrict__ err_rows,
                          int* __restrict__ done) {
  __shared__ float sh[256];
  float acc = 0.0f;
  for (int i = threadIdx.x; i < B_ * N_; i += 256) acc += err_rows[i];
  sh[threadIdx.x] = acc;
  __syncthreads();
  for (int off = 128; off; off >>= 1) {
    if (threadIdx.x < off) sh[threadIdx.x] += sh[threadIdx.x + off];
    __syncthreads();
  }
  if (threadIdx.x == 0 && (sh[0] / B_) < THRESH_) *done = 1;
}

// ------------------------ final cost: row partials -------------------------
__global__ void cost_partial(const float* __restrict__ C,
                             const float* __restrict__ u,
                             const float* __restrict__ v,
                             float* __restrict__ partial) {
  const int wid = threadIdx.x >> 5, lane = threadIdx.x & 31;
  const int row = blockIdx.x * 8 + wid;  // b*N + n
  const int b = row >> 10;
  const float* Crow = C + (size_t)row * M_;
  const float* vb = v + b * M_;
  const float uv = u[row];
  float acc = 0.0f;
  for (int j = lane; j < M_; j += 32) {
    float c = Crow[j];
    acc += expf((uv + vb[j] - c) * INV_EPS_) * c;
  }
  for (int off = 16; off; off >>= 1) acc += __shfl_xor(acc, off, 32);
  if (lane == 0) partial[row] = acc;
}

__global__ void cost_final(const float* __restrict__ partial,
                           float* __restrict__ out) {
  const int b = blockIdx.x;
  __shared__ float sh[256];
  float acc = 0.0f;
  for (int i = threadIdx.x; i < N_; i += 256) acc += partial[b * N_ + i];
  sh[threadIdx.x] = acc;
  __syncthreads();
  for (int off = 128; off; off >>= 1) {
    if (threadIdx.x < off) sh[threadIdx.x] += sh[threadIdx.x + off];
    __syncthreads();
  }
  if (threadIdx.x == 0) out[b] = sh[0];
}

// --------------------------------- launch ----------------------------------
extern "C" void kernel_launch(void* const* d_in, const int* in_sizes, int n_in,
                              void* d_out, int out_size, void* d_ws,
                              size_t ws_size, hipStream_t stream) {
  (void)in_sizes; (void)n_in; (void)out_size; (void)ws_size;
  const float* x  = (const float*)d_in[0];
  const float* y  = (const float*)d_in[1];
  const float* nu = (const float*)d_in[2];
  float* out = (float*)d_out;

  char* ws = (char*)d_ws;
  size_t off = 0;
  auto alloc = [&](size_t bytes) -> void* {
    void* p = ws + off;
    off = (off + bytes + 255) & ~(size_t)255;
    return p;
  };
  _Float16* Xh = (_Float16*)alloc((size_t)B_ * N_ * D_ * sizeof(_Float16));
  _Float16* Yh = (_Float16*)alloc((size_t)B_ * M_ * D_ * sizeof(_Float16));
  float* nx       = (float*)alloc((size_t)B_ * N_ * sizeof(float));
  float* ny       = (float*)alloc((size_t)B_ * M_ * sizeof(float));
  float* C        = (float*)alloc((size_t)B_ * N_ * M_ * sizeof(float));
  float* CT       = (float*)alloc((size_t)B_ * M_ * N_ * sizeof(float));
  float* u        = (float*)alloc((size_t)B_ * N_ * sizeof(float));
  float* v        = (float*)alloc((size_t)B_ * M_ * sizeof(float));
  float* err_rows = (float*)alloc((size_t)B_ * N_ * sizeof(float));
  float* partial  = (float*)alloc((size_t)B_ * N_ * sizeof(float));
  int* done       = (int*)alloc(sizeof(int));

  init_state<<<(B_ * N_ + 255) / 256, 256, 0, stream>>>(u, v, done);
  norm_convert<<<(B_ * N_) / 8, 256, 0, stream>>>(x, Xh, nx, B_ * N_);
  norm_convert<<<(B_ * M_) / 8, 256, 0, stream>>>(y, Yh, ny, B_ * M_);

  dim3 ggrid((N_ / 64) * (M_ / 64), B_);
  gemm_cost<<<ggrid, 256, 0, stream>>>(Xh, Yh, nx, ny, C, CT);

  for (int it = 0; it < MAX_ITER_; ++it) {
    sink_u<<<(B_ * N_) / 8, 256, 0, stream>>>(C, v, u, err_rows, done);
    sink_v<<<(B_ * M_) / 8, 256, 0, stream>>>(CT, u, nu, v, done);
    sink_done<<<1, 256, 0, stream>>>(err_rows, done);
  }

  cost_partial<<<(B_ * N_) / 8, 256, 0, stream>>>(C, u, v, partial);
  cost_final<<<B_, 256, 0, stream>>>(partial, out);
}
